// LinformerSelfAttention_10419590660281
// MI455X (gfx1250) — matrix operand order repaired and checked by reading, not compiled, not run on hardware
//
#include <hip/hip_runtime.h>
#include <hip/hip_bf16.h>

// ---------------- types ----------------
typedef __bf16 bf16;
typedef __attribute__((ext_vector_type(4)))  __bf16 bf16x4;
typedef __attribute__((ext_vector_type(8)))  __bf16 bf16x8;
typedef __attribute__((ext_vector_type(16))) __bf16 v16bf;
typedef __attribute__((ext_vector_type(8)))  float  v8f;
typedef __attribute__((ext_vector_type(4)))  int    v4i;

// async global->LDS path (gfx1250), guarded so either toolchain compiles
#if defined(__has_builtin)
#  if __has_builtin(__builtin_amdgcn_global_load_async_to_lds_b128)
#    define HAVE_ASYNC_LDS 1
#  endif
#endif
#ifndef HAVE_ASYNC_LDS
#  define HAVE_ASYNC_LDS 0
#endif

#if HAVE_ASYNC_LDS
// 16B per lane, memory -> LDS, tracked by ASYNCcnt (no VGPR round-trip).
// ROCm 7.2 signature: (v4i addrspace(1)*, v4i addrspace(3)*, imm offset, imm cpol)
__device__ __forceinline__ void async_copy16(const void* g, void* l) {
  __attribute__((address_space(1))) void* gp =
      (__attribute__((address_space(1))) void*)(void*)g;
  __attribute__((address_space(3))) void* lp =
      (__attribute__((address_space(3))) void*)l;
  __builtin_amdgcn_global_load_async_to_lds_b128(
      (__attribute__((address_space(1))) v4i*)gp,
      (__attribute__((address_space(3))) v4i*)lp, 0, 0);
}
#endif

__device__ __forceinline__ bf16 f2bf(float f) {
  unsigned u = __builtin_bit_cast(unsigned, f);
  unsigned r = u + 0x7FFFu + ((u >> 16) & 1u);      // round-to-nearest-even
  unsigned short s = (unsigned short)(r >> 16);
  return __builtin_bit_cast(bf16, s);
}

// ---------------- fp32 -> bf16 convert ----------------
__global__ __launch_bounds__(256) void cvt_f32_bf16(const float* __restrict__ src,
                                                    bf16* __restrict__ dst, int n) {
  int i4 = (blockIdx.x * 256 + threadIdx.x) * 4;
  if (i4 < n) {
    float4 v = *(const float4*)(src + i4);
    bf16x4 o;
    o[0] = f2bf(v.x); o[1] = f2bf(v.y); o[2] = f2bf(v.z); o[3] = f2bf(v.w);
    *(bf16x4*)(dst + i4) = o;
  }
}

// ---------------- generic bf16 WMMA GEMM ----------------
// C[M,N] = op(A) * B ;  A row-major [M,K] (or [K,M] when TRANSA), B row-major [K,N]
// Block tile 128x128, 8 waves (4 M-waves x 2 N-waves), each wave 32x64 = 2x4 WMMA tiles.
#define BM 128
#define BN 128
#define KT 32
#define APITCH 40   // halves; 80B rows -> 16B-aligned b128 fragment loads
#define BPITCH 40

template <bool TRANSA>
__global__ __launch_bounds__(256) void gemm_bf16(const bf16* __restrict__ A,
                                                 const bf16* __restrict__ Bm,
                                                 bf16* __restrict__ C,
                                                 int M, int N, int K,
                                                 long sA, long sB, long sC) {
  __shared__ bf16 As[BM * APITCH];
  __shared__ bf16 Bs[BN * BPITCH];

  const int bz = blockIdx.z;
  A  += (long)bz * sA;
  Bm += (long)bz * sB;
  C  += (long)bz * sC;

  const int tid  = threadIdx.x;
  const int lane = tid & 31;
  const int w    = tid >> 5;
  const int mw   = w & 3;     // 0..3
  const int nw   = w >> 2;    // 0..1
  const int lr   = lane & 15;
  const int lh   = lane >> 4;

  const int n0 = blockIdx.x * BN;
  const int m0 = blockIdx.y * BM;

  bf16x8 ra[2], rb[2];

  auto loadTile = [&](int k0) {
    if (!TRANSA) {
#pragma unroll
      for (int i = 0; i < 2; ++i) {               // 128 rows x 32 halves = 512 chunks
        int c = tid * 2 + i;
        int r = c >> 2, cc = (c & 3) << 3;
        ra[i] = *(const bf16x8*)&A[(long)(m0 + r) * K + k0 + cc];
      }
    } else {
#pragma unroll
      for (int i = 0; i < 2; ++i) {               // 32 rows x 128 halves
        int c = tid * 2 + i;
        int kk = c >> 4, mm = (c & 15) << 3;
        ra[i] = *(const bf16x8*)&A[(long)(k0 + kk) * M + m0 + mm];
      }
    }
#pragma unroll
    for (int i = 0; i < 2; ++i) {                 // B: 32 rows x 128 halves
      int c = tid * 2 + i;
      int kk = c >> 4, nn = (c & 15) << 3;
      rb[i] = *(const bf16x8*)&Bm[(long)(k0 + kk) * N + n0 + nn];
    }
  };

  auto stageLDS = [&]() {
    if (!TRANSA) {
#pragma unroll
      for (int i = 0; i < 2; ++i) {
        int c = tid * 2 + i;
        int r = c >> 2, cc = (c & 3) << 3;
        *(bf16x8*)&As[r * APITCH + cc] = ra[i];
      }
    } else {
#pragma unroll
      for (int i = 0; i < 2; ++i) {
        int c = tid * 2 + i;
        int kk = c >> 4, mm = (c & 15) << 3;
#pragma unroll
        for (int j = 0; j < 8; ++j) As[(mm + j) * APITCH + kk] = ra[i][j];
      }
    }
#pragma unroll
    for (int i = 0; i < 2; ++i) {                 // store B transposed: Bs[n][kk]
      int c = tid * 2 + i;
      int kk = c >> 4, nn = (c & 15) << 3;
#pragma unroll
      for (int j = 0; j < 8; ++j) Bs[(nn + j) * BPITCH + kk] = rb[i][j];
    }
  };

  v8f acc[2][4];
#pragma unroll
  for (int mi = 0; mi < 2; ++mi)
#pragma unroll
    for (int ni = 0; ni < 4; ++ni) acc[mi][ni] = (v8f){0, 0, 0, 0, 0, 0, 0, 0};

  loadTile(0);
  stageLDS();
  const int nk = K / KT;
  for (int kt = 0; kt < nk; ++kt) {
    __syncthreads();
    if (kt + 1 < nk) loadTile((kt + 1) * KT);

    // pull tile kt+2 toward cache while computing (global_prefetch_b8)
    if (kt + 2 < nk) {
      int k2 = (kt + 2) * KT;
      int c = tid * 2;
      if (!TRANSA)
        __builtin_prefetch(&A[(long)(m0 + (c >> 2)) * K + k2 + ((c & 3) << 3)], 0, 0);
      else
        __builtin_prefetch(&A[(long)(k2 + (c >> 4)) * M + m0 + ((c & 15) << 3)], 0, 0);
      __builtin_prefetch(&Bm[(long)(k2 + (c >> 4)) * N + n0 + ((c & 15) << 3)], 0, 0);
    }

    union F16 { v16bf v; bf16x8 h[2]; };
    F16 af[2], bfrag[4];
#pragma unroll
    for (int mi = 0; mi < 2; ++mi) {
      int r = mw * 32 + mi * 16 + lr;
      af[mi].h[0] = *(const bf16x8*)&As[r * APITCH + lh * 8];
      af[mi].h[1] = *(const bf16x8*)&As[r * APITCH + 16 + lh * 8];
    }
#pragma unroll
    for (int ni = 0; ni < 4; ++ni) {
      int nn = nw * 64 + ni * 16 + lr;
      bfrag[ni].h[0] = *(const bf16x8*)&Bs[nn * BPITCH + lh * 8];
      bfrag[ni].h[1] = *(const bf16x8*)&Bs[nn * BPITCH + 16 + lh * 8];
    }
#pragma unroll
    for (int mi = 0; mi < 2; ++mi)
#pragma unroll
      for (int ni = 0; ni < 4; ++ni)
        acc[mi][ni] = __builtin_amdgcn_wmma_f32_16x16x32_bf16(
            false, af[mi].v, false, bfrag[ni].v, (short)0, acc[mi][ni], false, false);

    __syncthreads();
    if (kt + 1 < nk) stageLDS();
  }

#pragma unroll
  for (int mi = 0; mi < 2; ++mi)
#pragma unroll
    for (int ni = 0; ni < 4; ++ni)
#pragma unroll
      for (int v = 0; v < 8; ++v) {
        int row = m0 + mw * 32 + mi * 16 + v + 8 * lh;
        int col = n0 + nw * 64 + ni * 16 + lr;
        C[(long)row * N + col] = f2bf(acc[mi][ni][v]);
      }
}

// ---------------- fused Linformer attention ----------------
// grid (n/128, h, b), 256 threads. Wave w owns query rows [blk*128+16w, +16).
// scores[16,256] = Q[16,64] * Klow^T  -> softmax -> out[16,64] = P[16,256] * Vlow
#define KSP 72    // Klow LDS pitch (halves)
#define VSP 264   // Vlow^T LDS pitch
#define PSP 264   // P LDS pitch

__global__ __launch_bounds__(256) void linformer_attn(const bf16* __restrict__ Q,
                                                      const bf16* __restrict__ Kl,
                                                      const bf16* __restrict__ Vl,
                                                      float* __restrict__ O,
                                                      int n, int d, int kdim) {
  extern __shared__ char smem_raw[];
  bf16* Ks = (bf16*)smem_raw;            // [256][KSP]   Klow rows (dh-contiguous)
  bf16* Vs = Ks + 256 * KSP;             // [64][VSP]    Vlow transposed (k-contiguous)
  bf16* Ps = Vs + 64 * VSP;              // 8 x [16][PSP]

  const int blk = blockIdx.x, h = blockIdx.y, b = blockIdx.z;
  const int tid = threadIdx.x, lane = tid & 31, w = tid >> 5;
  const int lr = lane & 15, lh = lane >> 4;
  const int dh = 64;

  const long qbase = ((long)b * n) * d + h * dh;
  const long kbase = ((long)b * kdim) * d + h * dh;

  // ---- stage Klow [256 x 64] (identity copy -> async DMA to LDS if available) ----
#if HAVE_ASYNC_LDS
#pragma unroll
  for (int i = 0; i < 8; ++i) {
    int c = i * 256 + tid;
    int r = c >> 3, cc = (c & 7) << 3;
    async_copy16(&Kl[kbase + (long)r * d + cc], &Ks[r * KSP + cc]);
  }
#else
#pragma unroll
  for (int i = 0; i < 8; ++i) {
    int c = i * 256 + tid;
    int r = c >> 3, cc = (c & 7) << 3;
    bf16x8 v = *(const bf16x8*)&Kl[kbase + (long)r * d + cc];
    *(bf16x8*)&Ks[r * KSP + cc] = v;
  }
#endif
  // ---- stage Vlow^T [64 x 256] (transpose -> manual scatter) ----
#pragma unroll
  for (int i = 0; i < 8; ++i) {
    int c = i * 256 + tid;
    int r = c >> 3, cc = (c & 7) << 3;
    bf16x8 v = *(const bf16x8*)&Vl[kbase + (long)r * d + cc];
#pragma unroll
    for (int j = 0; j < 8; ++j) Vs[(cc + j) * VSP + r] = v[j];
  }
#if HAVE_ASYNC_LDS
#  if __has_builtin(__builtin_amdgcn_s_wait_asynccnt)
  __builtin_amdgcn_s_wait_asynccnt(0);
#  else
  asm volatile("s_wait_asynccnt 0" ::: "memory");
#  endif
#endif
  __syncthreads();

  union F16 { v16bf v; bf16x8 h[2]; };

  // Q fragments (A-matrix layout): row = lr, halves per ISA 16-bit A layout
  const int row0 = blk * 128 + w * 16;
  const bf16* qrow = Q + qbase + (long)(row0 + lr) * d;
  F16 qf[2];
#pragma unroll
  for (int s = 0; s < 2; ++s) {
    qf[s].h[0] = *(const bf16x8*)&qrow[s * 32 + lh * 8];
    qf[s].h[1] = *(const bf16x8*)&qrow[s * 32 + 16 + lh * 8];
  }

  // ---- scores: 16 N-tiles x 2 K-steps = 32 WMMAs, double-buffered B-frags ----
  F16 kb[2][2];
  auto loadKB = [&](int t, int buf) {
    int kcol = t * 16 + lr;
#pragma unroll
    for (int s = 0; s < 2; ++s) {
      kb[buf][s].h[0] = *(const bf16x8*)&Ks[kcol * KSP + s * 32 + lh * 8];
      kb[buf][s].h[1] = *(const bf16x8*)&Ks[kcol * KSP + s * 32 + 16 + lh * 8];
    }
  };
  loadKB(0, 0);
  v8f sc[16];
#pragma unroll
  for (int t = 0; t < 16; ++t) {
    if (t + 1 < 16) loadKB(t + 1, (t + 1) & 1);   // prefetch next tile's frags
    v8f a = (v8f){0, 0, 0, 0, 0, 0, 0, 0};
#pragma unroll
    for (int s = 0; s < 2; ++s)
      a = __builtin_amdgcn_wmma_f32_16x16x32_bf16(false, qf[s].v, false, kb[t & 1][s].v,
                                                  (short)0, a, false, false);
    sc[t] = a;
  }

  // ---- softmax: C-layout row = v + 8*lh, col = t*16 + lr ----
  float rmax[8], rinv[8];
#pragma unroll
  for (int v = 0; v < 8; ++v) {
    float m = sc[0][v];
#pragma unroll
    for (int t = 1; t < 16; ++t) m = fmaxf(m, sc[t][v]);
#pragma unroll
    for (int off = 1; off < 16; off <<= 1) m = fmaxf(m, __shfl_xor(m, off, 32));
    rmax[v] = m;
  }
  bf16* pw = Ps + w * 16 * PSP;
  float rsum[8];
#pragma unroll
  for (int v = 0; v < 8; ++v) rsum[v] = 0.f;
#pragma unroll
  for (int t = 0; t < 16; ++t)
#pragma unroll
    for (int v = 0; v < 8; ++v) {
      float p = __expf(sc[t][v] - rmax[v]);
      rsum[v] += p;
      pw[(v + 8 * lh) * PSP + t * 16 + lr] = f2bf(p);   // unnormalized P
    }
#pragma unroll
  for (int v = 0; v < 8; ++v) {
    float s = rsum[v];
#pragma unroll
    for (int off = 1; off < 16; off <<= 1) s += __shfl_xor(s, off, 32);
    rinv[v] = 1.0f / s;
  }
  asm volatile("s_wait_dscnt 0" ::: "memory");  // P stores visible to own-wave loads

  // ---- out = P * Vlow : 8 K-steps x 4 N-tiles = 32 WMMAs, double-buffered frags ----
  F16 pa[2], vb[2][4];
  auto loadPV = [&](int s2, int buf) {
    pa[buf].h[0] = *(const bf16x8*)&pw[lr * PSP + s2 * 32 + lh * 8];
    pa[buf].h[1] = *(const bf16x8*)&pw[lr * PSP + s2 * 32 + 16 + lh * 8];
#pragma unroll
    for (int ni = 0; ni < 4; ++ni) {
      int dcol = ni * 16 + lr;
      vb[buf][ni].h[0] = *(const bf16x8*)&Vs[dcol * VSP + s2 * 32 + lh * 8];
      vb[buf][ni].h[1] = *(const bf16x8*)&Vs[dcol * VSP + s2 * 32 + 16 + lh * 8];
    }
  };
  loadPV(0, 0);
  v8f oacc[4];
#pragma unroll
  for (int ni = 0; ni < 4; ++ni) oacc[ni] = (v8f){0, 0, 0, 0, 0, 0, 0, 0};
#pragma unroll
  for (int s2 = 0; s2 < 8; ++s2) {
    if (s2 + 1 < 8) loadPV(s2 + 1, (s2 + 1) & 1);
#pragma unroll
    for (int ni = 0; ni < 4; ++ni)
      oacc[ni] = __builtin_amdgcn_wmma_f32_16x16x32_bf16(false, pa[s2 & 1].v, false,
                                                         vb[s2 & 1][ni].v, (short)0,
                                                         oacc[ni], false, false);
  }

  // fold 1/rowsum into fp32 epilogue; out[b, row, h*dh + col]
#pragma unroll
  for (int ni = 0; ni < 4; ++ni)
#pragma unroll
    for (int v = 0; v < 8; ++v) {
      long row = row0 + v + 8 * lh;
      O[((long)b * n + row) * d + h * dh + ni * 16 + lr] = oacc[ni][v] * rinv[v];
    }
}

// ---------------- host launcher ----------------
extern "C" void kernel_launch(void* const* d_in, const int* in_sizes, int n_in,
                              void* d_out, int out_size, void* d_ws, size_t ws_size,
                              hipStream_t stream) {
  const float* x  = (const float*)d_in[0];
  const float* Wq = (const float*)d_in[1];
  const float* Wk = (const float*)d_in[2];
  const float* Wv = (const float*)d_in[3];
  const float* pk = (const float*)d_in[4];
  const float* pv = (const float*)d_in[5];
  float* out = (float*)d_out;

  const int b = 4, n = 4096, d = 1024, kk = 256;
  const int h = 16;

  char* ws = (char*)d_ws;
  bf16* xh  = (bf16*)ws; ws += (size_t)b * n * d * 2;
  bf16* wqh = (bf16*)ws; ws += (size_t)d * d * 2;
  bf16* wkh = (bf16*)ws; ws += (size_t)d * d * 2;
  bf16* wvh = (bf16*)ws; ws += (size_t)d * d * 2;
  bf16* pkh = (bf16*)ws; ws += (size_t)n * kk * 2;
  bf16* pvh = (bf16*)ws; ws += (size_t)n * kk * 2;
  bf16* qh  = (bf16*)ws; ws += (size_t)b * n * d * 2;
  bf16* kvh = (bf16*)ws; ws += (size_t)b * n * d * 2;   // reused for keys then vals
  bf16* klh = (bf16*)ws; ws += (size_t)b * kk * d * 2;
  bf16* vlh = (bf16*)ws; ws += (size_t)b * kk * d * 2;

  auto cvt = [&](const float* s, bf16* dd, int cnt) {
    cvt_f32_bf16<<<(cnt / 4 + 255) / 256, 256, 0, stream>>>(s, dd, cnt);
  };
  cvt(x, xh, b * n * d);
  cvt(Wq, wqh, d * d);
  cvt(Wk, wkh, d * d);
  cvt(Wv, wvh, d * d);
  cvt(pk, pkh, n * kk);
  cvt(pv, pvh, n * kk);

  dim3 blkDim(256);
  // Q = X * Wq            [16384 x 1024] * [1024 x 1024]
  gemm_bf16<false><<<dim3(d / BN, (b * n) / BM, 1), blkDim, 0, stream>>>(
      xh, wqh, qh, b * n, d, d, 0, 0, 0);
  // keys = X * Wk
  gemm_bf16<false><<<dim3(d / BN, (b * n) / BM, 1), blkDim, 0, stream>>>(
      xh, wkh, kvh, b * n, d, d, 0, 0, 0);
  // Klow[b] = proj_k^T * keys[b]   [256 x 4096] * [4096 x 1024]
  gemm_bf16<true><<<dim3(d / BN, kk / BM, b), blkDim, 0, stream>>>(
      pkh, kvh, klh, kk, d, n, 0, (long)n * d, (long)kk * d);
  // vals = X * Wv  (reuse kvh)
  gemm_bf16<false><<<dim3(d / BN, (b * n) / BM, 1), blkDim, 0, stream>>>(
      xh, wvh, kvh, b * n, d, d, 0, 0, 0);
  // Vlow[b] = proj_v^T * vals[b]
  gemm_bf16<true><<<dim3(d / BN, kk / BM, b), blkDim, 0, stream>>>(
      pvh, kvh, vlh, kk, d, n, 0, (long)n * d, (long)kk * d);

  // fused attention
  size_t smem = (size_t)(256 * KSP + 64 * VSP + 8 * 16 * PSP) * sizeof(bf16);
  linformer_attn<<<dim3(n / 128, h, b), blkDim, smem, stream>>>(qh, klh, vlh, out, n, d, kk);
}
